// EncoderLayer_54142357733581
// MI455X (gfx1250) — matrix operand order repaired and checked by reading, not compiled, not run on hardware
//
#include <hip/hip_runtime.h>
#include <cmath>
#include <cstdint>

// ---------------------------------------------------------------------------
// x[S=50, L=1024, D=768], D4=192. ~600 GFLOP of batched GEMMs on
// v_wmma_f32_16x16x32_f16 (f16 in, f32 accumulate). fp32 softmax/layernorm.
// ---------------------------------------------------------------------------
#define S_STEPS 50
#define LSEQ    1024
#define DMODEL  768
#define DBOT    192

typedef __attribute__((ext_vector_type(16))) _Float16 v16h;
typedef __attribute__((ext_vector_type(8)))  float    v8f;
typedef __attribute__((ext_vector_type(4)))  float    f32x4;
typedef __attribute__((ext_vector_type(2)))  _Float16 h2;
typedef __attribute__((ext_vector_type(2)))  __fp16   fp16x2;

union FragAB { v16h v; unsigned u[8]; };

// LDS strides in halves; even -> every fragment read is >=4-byte aligned.
#define LDSA_STR 40   // A tile: 128 rows x 32 K (+8 pad)
#define LDSB_STR 40   // B tile stored transposed: 64 cols x 32 K (+8 pad)

__device__ __forceinline__ h2 pack2(float x, float y) {
    // single v_cvt_pk_rtz_f16_f32 -> one 32-bit value -> ds_store_b32
    union { fp16x2 f; h2 h; } u;
    u.f = __builtin_amdgcn_cvt_pkrtz(x, y);
    return u.h;
}

// ---------------------------------------------------------------------------
// Strided-batched GEMM:
//   C[b,m,n] = alpha * sum_k A[b](m,k)*B[b](k,n) (+bias[b,n]) (relu?)
//   A(m,k) = A[b*sA + m*lda + k]
//   B(k,n) = transB ? B[b*sB + n*ldb + k] : B[b*sB + k*ldb + n]
// 256 thr = 8 waves; WG tile 128(M) x 64(N); wave tile 32x32 (2x2 WMMA).
// GUARD=false: dims tile-aligned -> float4 loads, no bounds checks.
// GUARD=true : branchless clamped loads + cndmask select (no exec waits).
// ---------------------------------------------------------------------------
template <bool GUARD>
__global__ __launch_bounds__(256)
void gemm_wmma_f16(const float* __restrict__ A, long long sA, int lda,
                   const float* __restrict__ B, long long sB, int ldb, int transB,
                   const float* __restrict__ bias, long long sBias,
                   float* __restrict__ C, long long sC, int ldc,
                   int M, int N, int K, float alpha, int relu)
{
    __shared__ _Float16 As[128 * LDSA_STR];
    __shared__ _Float16 Bs[64 * LDSB_STR];

    const int tid  = threadIdx.x;
    const int lane = tid & 31;          // wave32
    const int wave = tid >> 5;
    const int wm   = wave & 3;          // M subtile (x32)
    const int wn   = wave >> 2;         // N subtile (x32)
    const int bm   = blockIdx.y * 128;
    const int bn   = blockIdx.x * 64;
    const int bz   = blockIdx.z;

    A += (long long)bz * sA;
    B += (long long)bz * sB;
    C += (long long)bz * sC;
    const float* biasP = bias ? (bias + (long long)bz * sBias) : nullptr;

    v8f zero = {0.f, 0.f, 0.f, 0.f, 0.f, 0.f, 0.f, 0.f};
    v8f acc[2][2];
    acc[0][0] = zero; acc[0][1] = zero; acc[1][0] = zero; acc[1][1] = zero;

    // ---- register staging (software pipeline over K tiles) ----
    f32x4 aR4[4], bR4[2];          // fast path
    float aRs[16], bRs[8];         // guarded path

    auto loadGlobal = [&](int kt) {
        if constexpr (!GUARD) {
            #pragma unroll
            for (int i = 0; i < 4; ++i) {
                int f4 = tid + i * 256;            // 0..1023
                int r = f4 >> 3, kq = f4 & 7;      // row, K/4
                aR4[i] = *(const f32x4*)(A + (long long)(bm + r) * lda + kt + 4 * kq);
            }
            #pragma unroll
            for (int i = 0; i < 2; ++i) {
                int f4 = tid + i * 256;            // 0..511
                if (transB) {
                    int n = f4 >> 3, kq = f4 & 7;
                    bR4[i] = *(const f32x4*)(B + (long long)(bn + n) * ldb + kt + 4 * kq);
                } else {
                    int kr = f4 >> 4, nq = f4 & 15;
                    bR4[i] = *(const f32x4*)(B + (long long)(kt + kr) * ldb + bn + 4 * nq);
                }
            }
        } else {
            #pragma unroll
            for (int i = 0; i < 16; ++i) {
                int idx = tid + i * 256;           // 0..4095
                int r = idx >> 5, k = idx & 31;
                int gm = bm + r, gk = kt + k;
                int cm = gm < M ? gm : M - 1;      // clamped: always legal
                int ck = gk < K ? gk : K - 1;
                float v = A[(long long)cm * lda + ck];
                aRs[i] = (gm < M && gk < K) ? v : 0.f;
            }
            #pragma unroll
            for (int i = 0; i < 8; ++i) {
                int idx = tid + i * 256;           // 0..2047
                int k, n;
                if (transB) { n = idx >> 5; k = idx & 31; }
                else        { k = idx >> 6; n = idx & 63; }
                int gk = kt + k, gn = bn + n;
                int cn = gn < N ? gn : N - 1;
                int ck = gk < K ? gk : K - 1;
                float v = transB ? B[(long long)cn * ldb + ck]
                                 : B[(long long)ck * ldb + cn];
                bRs[i] = (gk < K && gn < N) ? v : 0.f;
            }
        }
    };

    auto storeLDS = [&]() {
        if constexpr (!GUARD) {
            #pragma unroll
            for (int i = 0; i < 4; ++i) {
                int f4 = tid + i * 256;
                int r = f4 >> 3, k = (f4 & 7) * 4;
                *(h2*)(&As[r * LDSA_STR + k])     = pack2(aR4[i].x, aR4[i].y);
                *(h2*)(&As[r * LDSA_STR + k + 2]) = pack2(aR4[i].z, aR4[i].w);
            }
            #pragma unroll
            for (int i = 0; i < 2; ++i) {
                int f4 = tid + i * 256;
                if (transB) {
                    int n = f4 >> 3, k = (f4 & 7) * 4;
                    *(h2*)(&Bs[n * LDSB_STR + k])     = pack2(bR4[i].x, bR4[i].y);
                    *(h2*)(&Bs[n * LDSB_STR + k + 2]) = pack2(bR4[i].z, bR4[i].w);
                } else {
                    int kr = f4 >> 4, n = (f4 & 15) * 4;   // store transposed
                    Bs[(n + 0) * LDSB_STR + kr] = (_Float16)bR4[i].x;
                    Bs[(n + 1) * LDSB_STR + kr] = (_Float16)bR4[i].y;
                    Bs[(n + 2) * LDSB_STR + kr] = (_Float16)bR4[i].z;
                    Bs[(n + 3) * LDSB_STR + kr] = (_Float16)bR4[i].w;
                }
            }
        } else {
            #pragma unroll
            for (int i = 0; i < 16; ++i) {
                int idx = tid + i * 256;
                int r = idx >> 5, k = idx & 31;
                As[r * LDSA_STR + k] = (_Float16)aRs[i];
            }
            #pragma unroll
            for (int i = 0; i < 8; ++i) {
                int idx = tid + i * 256;
                int k, n;
                if (transB) { n = idx >> 5; k = idx & 31; }
                else        { k = idx >> 6; n = idx & 63; }
                Bs[n * LDSB_STR + k] = (_Float16)bRs[i];
            }
        }
    };

    loadGlobal(0);
    for (int kt = 0; kt < K; kt += 32) {
        storeLDS();
        __syncthreads();
        if (kt + 32 < K) loadGlobal(kt + 32);      // overlap with WMMA below

        // ---- fragments in contiguous K order: k = 16*(lane/16) + s, same in A and B ----
        FragAB afr[2], bfr[2];
        const int khiA = (lane >= 16) ? 16 : 0;
        #pragma unroll
        for (int s = 0; s < 2; ++s) {
            int row = wm * 32 + s * 16 + (lane & 15);
            const _Float16* p = &As[row * LDSA_STR];
            #pragma unroll
            for (int v = 0; v < 8; ++v) {
                int k0 = 2 * v + khiA;
                afr[s].u[v] = *(const unsigned*)(p + k0);   // K pair (k0,k0+1)
            }
        }
        const int kbB = (lane >= 16) ? 16 : 0;
        #pragma unroll
        for (int s = 0; s < 2; ++s) {
            int col = wn * 32 + s * 16 + (lane & 15);
            const _Float16* p = &Bs[col * LDSB_STR + kbB];
            #pragma unroll
            for (int v = 0; v < 8; ++v)
                bfr[s].u[v] = *(const unsigned*)(p + 2 * v);
        }
        #pragma unroll
        for (int i = 0; i < 2; ++i)
            #pragma unroll
            for (int j = 0; j < 2; ++j)
                acc[i][j] = __builtin_amdgcn_wmma_f32_16x16x32_f16(
                    false, afr[i].v, false, bfr[j].v,
                    (short)0, acc[i][j], false, false);
        __syncthreads();
    }

    // ---- store C (lane&15 = N, VGPR v + 8*(lane/16) = M) ----
    const int cn0 = (lane & 15);
    const int cm0 = (lane >> 4) << 3;
    #pragma unroll
    for (int i = 0; i < 2; ++i)
        #pragma unroll
        for (int j = 0; j < 2; ++j) {
            int n = bn + wn * 32 + j * 16 + cn0;
            float bv = 0.f;
            if (biasP) {
                int cn = (!GUARD || n < N) ? n : 0;
                bv = biasP[cn];
            }
            #pragma unroll
            for (int v = 0; v < 8; ++v) {
                int m = bm + wm * 32 + i * 16 + v + cm0;
                if (!GUARD || (m < M && n < N)) {
                    float val = acc[i][j][v] * alpha + bv;
                    if (relu) val = fmaxf(val, 0.f);
                    C[(long long)m * ldc + n] = val;
                }
            }
        }
}

// ---------------------------------------------------------------------------
// In-place row softmax; one block per contiguous row of length `len`.
// ---------------------------------------------------------------------------
__global__ __launch_bounds__(256)
void softmax_rows(float* __restrict__ data, int len)
{
    __shared__ float red[256];
    const int tid = threadIdx.x;
    float* p = data + (size_t)blockIdx.x * len;

    float m = -3.0e38f;
    for (int i = tid; i < len; i += 256) m = fmaxf(m, p[i]);
    red[tid] = m; __syncthreads();
    for (int s = 128; s > 0; s >>= 1) {
        if (tid < s) red[tid] = fmaxf(red[tid], red[tid + s]);
        __syncthreads();
    }
    const float mx = red[0]; __syncthreads();

    float sum = 0.f;
    for (int i = tid; i < len; i += 256) {
        float e = __expf(p[i] - mx); p[i] = e; sum += e;
    }
    red[tid] = sum; __syncthreads();
    for (int s = 128; s > 0; s >>= 1) {
        if (tid < s) red[tid] += red[tid + s];
        __syncthreads();
    }
    const float inv = 1.f / red[0];
    for (int i = tid; i < len; i += 256) p[i] *= inv;
}

// ---------------------------------------------------------------------------
// out = LayerNorm(x + h) * g + b ; gamma/beta row = (row / rowsPerStep)*gstride
// ---------------------------------------------------------------------------
__global__ __launch_bounds__(256)
void add_layernorm(const float* __restrict__ x, const float* __restrict__ h,
                   const float* __restrict__ g, const float* __restrict__ b,
                   float* __restrict__ out, int d, int gstride, int rowsPerStep)
{
    __shared__ float r1[256], r2[256];
    const int tid = threadIdx.x;
    const size_t row = blockIdx.x;
    const float* xr = x + row * d;
    const float* hr = h + row * d;
    float* orow = out + row * d;
    const int step = blockIdx.x / rowsPerStep;
    const float* gr = g + (size_t)step * gstride;
    const float* br = b + (size_t)step * gstride;

    float v[4];
    float s1 = 0.f, s2 = 0.f;
    int cnt = 0;
    for (int i = tid; i < d; i += 256) {
        float t = xr[i] + hr[i];
        v[cnt++] = t; s1 += t; s2 += t * t;
    }
    r1[tid] = s1; r2[tid] = s2; __syncthreads();
    for (int s = 128; s > 0; s >>= 1) {
        if (tid < s) { r1[tid] += r1[tid + s]; r2[tid] += r2[tid + s]; }
        __syncthreads();
    }
    const float mean = r1[0] / d;
    const float var  = r2[0] / d - mean * mean;
    const float inv  = rsqrtf(var + 1e-5f);
    cnt = 0;
    for (int i = tid; i < d; i += 256)
        orow[i] = (v[cnt++] - mean) * inv * gr[i] + br[i];
}

// ---------------------------------------------------------------------------
// Orchestration. Workspace (floats): X0,X1,H,T: S*L*D ; Sc: S*L*L ; Q,K: S*L*D4
// ---------------------------------------------------------------------------
extern "C" void kernel_launch(void* const* d_in, const int* in_sizes, int n_in,
                              void* d_out, int out_size, void* d_ws, size_t ws_size,
                              hipStream_t stream)
{
    (void)in_sizes; (void)n_in; (void)out_size; (void)ws_size;
    typedef long long ll;

    const float* x       = (const float*)d_in[0];
    const float* pre_Wq  = (const float*)d_in[1];
    const float* pre_Wk  = (const float*)d_in[2];
    const float* pre_Wv  = (const float*)d_in[3];
    const float* ln1_g   = (const float*)d_in[4];
    const float* ln1_b   = (const float*)d_in[5];
    const float* pre_W1  = (const float*)d_in[6];
    const float* pre_b1  = (const float*)d_in[7];
    const float* pre_W2  = (const float*)d_in[8];
    const float* pre_b2  = (const float*)d_in[9];
    const float* ln4p_g  = (const float*)d_in[10];
    const float* ln4p_b  = (const float*)d_in[11];
    const float* cross_Wq= (const float*)d_in[12];
    const float* cross_Wk= (const float*)d_in[13];
    const float* cross_Wv= (const float*)d_in[14];
    const float* ln2_g   = (const float*)d_in[15];
    const float* ln2_b   = (const float*)d_in[16];
    const float* post_Wq = (const float*)d_in[17];
    const float* post_Wk = (const float*)d_in[18];
    const float* post_Wv = (const float*)d_in[19];
    const float* ln3_g   = (const float*)d_in[20];
    const float* ln3_b   = (const float*)d_in[21];
    const float* post_W1 = (const float*)d_in[22];
    const float* post_b1 = (const float*)d_in[23];
    const float* post_W2 = (const float*)d_in[24];
    const float* post_b2 = (const float*)d_in[25];
    const float* ln4_g   = (const float*)d_in[26];
    const float* ln4_b   = (const float*)d_in[27];

    const size_t nXD = (size_t)S_STEPS * LSEQ * DMODEL;
    const size_t nXF = (size_t)S_STEPS * LSEQ * DBOT;
    const size_t nSS = (size_t)S_STEPS * LSEQ * LSEQ;
    float* X0 = (float*)d_ws;
    float* X1 = X0 + nXD;
    float* Hh = X1 + nXD;
    float* Tt = Hh + nXD;
    float* Sc = Tt + nXD;
    float* Qq = Sc + nSS;     // also reused as MLP hidden
    float* Kk = Qq + nXF;

    const float invsq = 1.0f / sqrtf((float)DBOT);

    auto gemm = [&](const float* A, ll sA, int lda,
                    const float* B, ll sB, int ldb, int transB,
                    const float* bias, ll sBias,
                    float* C, ll sC, int ldc,
                    int M, int N, int K, int batch, float alpha, int relu) {
        dim3 grid((N + 63) / 64, (M + 127) / 128, batch);
        bool guard = (M % 128) || (N % 64) || (K % 32);
        if (guard)
            gemm_wmma_f16<true><<<grid, dim3(256), 0, stream>>>(
                A, sA, lda, B, sB, ldb, transB, bias, sBias,
                C, sC, ldc, M, N, K, alpha, relu);
        else
            gemm_wmma_f16<false><<<grid, dim3(256), 0, stream>>>(
                A, sA, lda, B, sB, ldb, transB, bias, sBias,
                C, sC, ldc, M, N, K, alpha, relu);
    };
    auto softmax = [&](float* p, int rows, int len) {
        softmax_rows<<<rows, 256, 0, stream>>>(p, len);
    };
    auto lnadd = [&](const float* xi, const float* hh, const float* g,
                     const float* b, float* o, int gstride) {
        add_layernorm<<<S_STEPS * LSEQ, 256, 0, stream>>>(
            xi, hh, g, b, o, DMODEL, gstride, LSEQ);
    };

    // per-step self attention: xout = LN(xin + ((softmax(q k^T/s) xin) Wv))
    auto self_attn = [&](const float* xin, const float* Wq, const float* Wk,
                         const float* Wv, const float* g, const float* b,
                         float* xout) {
        gemm(xin, (ll)LSEQ * DMODEL, DMODEL, Wq, (ll)DMODEL * DBOT, DBOT, 0,
             nullptr, 0, Qq, (ll)LSEQ * DBOT, DBOT,
             LSEQ, DBOT, DMODEL, S_STEPS, 1.f, 0);
        gemm(xin, (ll)LSEQ * DMODEL, DMODEL, Wk, (ll)DMODEL * DBOT, DBOT, 0,
             nullptr, 0, Kk, (ll)LSEQ * DBOT, DBOT,
             LSEQ, DBOT, DMODEL, S_STEPS, 1.f, 0);
        gemm(Qq, (ll)LSEQ * DBOT, DBOT, Kk, (ll)LSEQ * DBOT, DBOT, 1,
             nullptr, 0, Sc, (ll)LSEQ * LSEQ, LSEQ,
             LSEQ, LSEQ, DBOT, S_STEPS, invsq, 0);
        softmax(Sc, S_STEPS * LSEQ, LSEQ);
        gemm(Sc, (ll)LSEQ * LSEQ, LSEQ, xin, (ll)LSEQ * DMODEL, DMODEL, 0,
             nullptr, 0, Tt, (ll)LSEQ * DMODEL, DMODEL,
             LSEQ, DMODEL, LSEQ, S_STEPS, 1.f, 0);
        gemm(Tt, (ll)LSEQ * DMODEL, DMODEL, Wv, (ll)DMODEL * DMODEL, DMODEL, 0,
             nullptr, 0, Hh, (ll)LSEQ * DMODEL, DMODEL,
             LSEQ, DMODEL, DMODEL, S_STEPS, 1.f, 0);
        lnadd(xin, Hh, g, b, xout, DMODEL);
    };

    // per-step bottleneck MLP: xout = LN(xin + relu(xin W1 + b1) W2 + b2)
    auto mlp = [&](const float* xin, const float* W1, const float* b1,
                   const float* W2, const float* b2, const float* g,
                   const float* bet, float* xout) {
        gemm(xin, (ll)LSEQ * DMODEL, DMODEL, W1, (ll)DMODEL * DBOT, DBOT, 0,
             b1, DBOT, Qq, (ll)LSEQ * DBOT, DBOT,
             LSEQ, DBOT, DMODEL, S_STEPS, 1.f, 1);
        gemm(Qq, (ll)LSEQ * DBOT, DBOT, W2, (ll)DBOT * DMODEL, DMODEL, 0,
             b2, DMODEL, Hh, (ll)LSEQ * DMODEL, DMODEL,
             LSEQ, DMODEL, DBOT, S_STEPS, 1.f, 0);
        lnadd(xin, Hh, g, bet, xout, DMODEL);
    };

    // ---- stage 1: pre self-attention (x -> X0) ----
    self_attn(x, pre_Wq, pre_Wk, pre_Wv, ln1_g, ln1_b, X0);
    // ---- stage 2: pre MLP (X0 -> X1) ----
    mlp(X0, pre_W1, pre_b1, pre_W2, pre_b2, ln4p_g, ln4p_b, X1);

    // ---- stage 3: cross-step attention over S, batched over l (X1 -> X0) ----
    gemm(X1, 0, DMODEL, cross_Wq, 0, DBOT, 0, nullptr, 0,
         Qq, 0, DBOT, S_STEPS * LSEQ, DBOT, DMODEL, 1, 1.f, 0);
    gemm(X1, 0, DMODEL, cross_Wk, 0, DBOT, 0, nullptr, 0,
         Kk, 0, DBOT, S_STEPS * LSEQ, DBOT, DMODEL, 1, 1.f, 0);
    // scores[l,i,j] = q[i,l,:].k[j,l,:] * invsq (batch=l via strides)
    gemm(Qq, (ll)DBOT, LSEQ * DBOT, Kk, (ll)DBOT, LSEQ * DBOT, 1,
         nullptr, 0, Sc, (ll)S_STEPS * S_STEPS, S_STEPS,
         S_STEPS, S_STEPS, DBOT, LSEQ, invsq, 0);
    softmax(Sc, LSEQ * S_STEPS, S_STEPS);
    // t[s,l,:] = sum_j a[l,s,j] * x[j,l,:]  (written in [S,L,D] layout)
    gemm(Sc, (ll)S_STEPS * S_STEPS, S_STEPS, X1, (ll)DMODEL, LSEQ * DMODEL, 0,
         nullptr, 0, Tt, (ll)DMODEL, LSEQ * DMODEL,
         S_STEPS, DMODEL, S_STEPS, LSEQ, 1.f, 0);
    gemm(Tt, 0, DMODEL, cross_Wv, 0, DMODEL, 0, nullptr, 0,
         Hh, 0, DMODEL, S_STEPS * LSEQ, DMODEL, DMODEL, 1, 1.f, 0);
    lnadd(X1, Hh, ln2_g, ln2_b, X0, 0);

    // ---- stage 4: post self-attention (X0 -> X1) ----
    self_attn(X0, post_Wq, post_Wk, post_Wv, ln3_g, ln3_b, X1);
    // ---- stage 5: post MLP (X1 -> d_out) ----
    mlp(X1, post_W1, post_b1, post_W2, post_b2, ln4_g, ln4_b, (float*)d_out);
}
